// MB_st_31714038513691
// MI455X (gfx1250) — matrix operand order repaired and checked by reading, not compiled, not run on hardware
//
#include <hip/hip_runtime.h>
#include <hip/hip_bf16.h>

typedef __attribute__((ext_vector_type(16))) _Float16 v16h;
typedef __attribute__((ext_vector_type(8)))  float    v8f;

#define Nn   16
#define Tt   8
#define Cc   256
#define HW   196
#define ICn  64
#define NT   128
#define CIN  257
#define KP   288   // padded cin (9 * 32)
#define NP   208   // padded hw  (13 * 16)
#define TILE 512   // halfs per 16x32 / 32x16 fragment tile (32 lanes * 16)
#define MASK_ELEMS (NT * 2 * 448 * 448)

// ---- fragment-major swizzles (wave32, contiguous k order used by this code for both A and B) ----
// A 16x32 f16: lanes0-15 row=lane: pos = K0-15; lanes16-31 row=lane-16: pos = K16-31
__device__ __forceinline__ int a_sw(int mr, int kr) {
  const int lane = mr + ((kr >> 4) << 4);
  const int pos  = kr & 15;
  return lane * 16 + pos;
}
// B 32x16 f16: lanes0-15 col=lane: pos = K0-15; lanes16-31 col=lane-16: pos = K16-31
__device__ __forceinline__ int b_sw(int kr, int nc) {
  const int lane = nc + ((kr >> 4) << 4);
  return lane * 16 + (kr & 15);
}

// ------------------- kernel 1: Q/K projections + V -------------------
__global__ void qkv_kernel(const float* __restrict__ x, const float* __restrict__ lamp,
                           const int* __restrict__ idxp, const float* __restrict__ Wq,
                           const float* __restrict__ bq, const float* __restrict__ Wv,
                           const float* __restrict__ bv,
                           _Float16* __restrict__ qt, _Float16* __restrict__ kt,
                           float* __restrict__ vbuf, float* __restrict__ vout) {
  extern __shared__ __align__(32) char smem[];
  _Float16* Alds = (_Float16*)smem;                          // [4 mt][9 kt][TILE]
  _Float16* Blds = (_Float16*)(smem + (size_t)ICn * KP * 2); // [9 kt][13 nt][TILE]

  const int n    = blockIdx.x;
  const int tid  = threadIdx.x;
  const int lane = tid & 31;
  const int wave = tid >> 5;
  const float lam = lamp[0] - 0.5f;

  // stage A = Wq (fragment-major, zero pad K to 288)
  for (int i = tid; i < ICn * KP; i += 256) {
    const int d = i / KP, c = i % KP;
    const _Float16 v = (c < CIN) ? (_Float16)Wq[d * CIN + c] : (_Float16)0.0f;
    Alds[((d >> 4) * 9 + (c >> 5)) * TILE + a_sw(d & 15, c & 31)] = v;
  }

  const int i0 = n / Tt, t0 = n % Tt;
  const int gshuf = idxp[i0] * Tt + t0;

  for (int pass = 0; pass < 2; ++pass) {
    const float* src   = x + (size_t)(pass ? gshuf : n) * Cc * HW;
    const float  lamch = pass ? -lam : lam;
    __builtin_prefetch(src, 0, 0);
    __syncthreads();
    // stage B = [x ; lam-channel] (fragment-major, zero pad to 288x208)
    for (int i = tid; i < KP * NP; i += 256) {
      const int k = i / NP, p = i % NP;
      float v = 0.0f;
      if (p < HW) {
        if (k < Cc)            v = src[k * HW + p];
        else if (k == CIN - 1) v = lamch;
      }
      Blds[((k >> 5) * 13 + (p >> 4)) * TILE + b_sw(k & 31, p & 15)] = (_Float16)v;
    }
    __syncthreads();

    const _Float16* abase = Alds + lane * 16;
    const _Float16* bbase = Blds + lane * 16;

    // 64x208 = A(64x288) * B(288x208); 4x13 tiles, 9 k-steps
    for (int tile = wave; tile < 4 * 13; tile += 8) {
      const int mt  = tile & 3;
      const int nt_ = tile >> 2;
      const int mrow = (mt << 4) + ((lane >> 4) << 3);
      v8f acc;
#pragma unroll
      for (int r = 0; r < 8; ++r) acc[r] = bq[mrow + r];   // bias in accumulator
#pragma unroll
      for (int ks = 0; ks < KP / 32; ++ks) {
        const v16h a = *(const v16h*)(abase + (mt * 9 + ks) * TILE);
        const v16h b = *(const v16h*)(bbase + (ks * 13 + nt_) * TILE);
        acc = __builtin_amdgcn_wmma_f32_16x16x32_f16(false, a, false, b,
                                                     (short)0, acc, false, false);
      }
      const int pcol = (nt_ << 4) + (lane & 15);
      // both Q and K stored transposed: [n][p][d], contiguous in d (b128 store)
      _Float16* dst = (pass == 0 ? qt : kt) + ((size_t)n * NP + pcol) * ICn + mrow;
#pragma unroll
      for (int r = 0; r < 8; ++r) dst[r] = (_Float16)acc[r];
    }

    if (pass == 1 && tid < HW) {             // V projection in fp32
      const int p = tid;
      float s = bv[0];
#pragma unroll 4
      for (int c = 0; c < Cc; ++c) s += Wv[c] * src[c * HW + p];
      s += Wv[CIN - 1] * (-lam);
      vbuf[n * HW + p] = s;
      vout[n * HW + p] = s;                  // second output (v_out)
    }
  }
}

// ---- kernel 2: scores = Qt*K/8, softmax+attn*v fused, sigmoid upsample ----
__global__ void attn_kernel(const _Float16* __restrict__ qt, const _Float16* __restrict__ kt,
                            const float* __restrict__ vbuf, float* __restrict__ out) {
  extern __shared__ __align__(32) char smem[];
  _Float16* Qlds = (_Float16*)smem;                    // [13 mt][2 kt][TILE]
  _Float16* Klds = Qlds + (size_t)NP * ICn;            // [2 kt][13 nt][TILE]
  float*    Slds = (float*)(Klds + (size_t)ICn * NP);  // [NP][NP]
  float*    Vlds = Slds + (size_t)NP * NP;             // [NP]
  float*    Sig  = Vlds + NP;                          // [NP]

  const int n    = blockIdx.x;
  const int tid  = threadIdx.x;
  const int lane = tid & 31;
  const int wave = tid >> 5;

  const _Float16* qsrc = qt + (size_t)n * NP * ICn;    // [p][d]
  const _Float16* ksrc = kt + (size_t)n * NP * ICn;    // [p][d]
  __builtin_prefetch(qsrc, 0, 0);
  __builtin_prefetch(ksrc, 0, 0);
  for (int i = tid; i < NP * ICn; i += 256) {
    const int p = i >> 6, d = i & 63;
    Qlds[((p >> 4) * 2 + (d >> 5)) * TILE + a_sw(p & 15, d & 31)] = qsrc[i];
    Klds[((d >> 5) * 13 + (p >> 4)) * TILE + b_sw(d & 31, p & 15)] = ksrc[i];
  }
  for (int i = tid; i < HW; i += 256) Vlds[i] = vbuf[n * HW + i];
  __syncthreads();

  const _Float16* abase = Qlds + lane * 16;
  const _Float16* bbase = Klds + lane * 16;

  // S(208x208) = Qt(208x64) * K(64x208), scaled by 1/sqrt(64)
  for (int tile = wave; tile < 13 * 13; tile += 8) {
    const int mt  = tile % 13;
    const int nt_ = tile / 13;
    v8f acc;
#pragma unroll
    for (int r = 0; r < 8; ++r) acc[r] = 0.0f;
#pragma unroll
    for (int ks = 0; ks < 2; ++ks) {
      const v16h a = *(const v16h*)(abase + (mt * 2 + ks) * TILE);
      const v16h b = *(const v16h*)(bbase + (ks * 13 + nt_) * TILE);
      acc = __builtin_amdgcn_wmma_f32_16x16x32_f16(false, a, false, b,
                                                   (short)0, acc, false, false);
    }
    const int mrow = (mt << 4) + ((lane >> 4) << 3);
    const int pc   = (nt_ << 4) + (lane & 15);
#pragma unroll
    for (int r = 0; r < 8; ++r) Slds[(mrow + r) * NP + pc] = acc[r] * 0.125f;
  }
  __syncthreads();

  // fused row-softmax + attn @ v + sigmoid (one row per thread)
  if (tid < HW) {
    const float* row = Slds + tid * NP;
    float mx = -3.402823466e38f;
    for (int p = 0; p < HW; ++p) mx = fmaxf(mx, row[p]);
    float sum = 0.0f, sv = 0.0f;
    for (int p = 0; p < HW; ++p) {
      const float e = __expf(row[p] - mx);
      sum += e;
      sv  += e * Vlds[p];
    }
    const float m = sv / sum;
    Sig[tid] = 1.0f / (1.0f + __expf(-m));
  }
  __syncthreads();

  // single-pass 32x nearest-upsample write: [2][448][448] fp32, float4 stores
  float4* ob = (float4*)out;
  for (int idx = tid; idx < 2 * 448 * 112; idx += 256) {
    const int xv = idx % 112;
    const int y  = (idx / 112) % 448;
    const int ch = idx / (112 * 448);
    const float s   = Sig[(y >> 5) * 14 + (xv >> 3)];
    const float val = ch ? s : 1.0f - s;
    const float4 o  = {val, val, val, val};
    ob[((size_t)(n * 2 + ch) * 448 + y) * 112 + xv] = o;
  }
}

extern "C" void kernel_launch(void* const* d_in, const int* in_sizes, int n_in,
                              void* d_out, int out_size, void* d_ws, size_t ws_size,
                              hipStream_t stream) {
  const float* x     = (const float*)d_in[0];
  const float* lam   = (const float*)d_in[1];
  const int*   index = (const int*)d_in[2];
  const float* Wq    = (const float*)d_in[3];
  const float* bq    = (const float*)d_in[4];
  const float* Wv    = (const float*)d_in[5];
  const float* bv    = (const float*)d_in[6];
  float* out = (float*)d_out;

  char* ws = (char*)d_ws;
  const size_t qk_bytes = (size_t)NT * NP * ICn * sizeof(_Float16);   // 3.4 MB each
  _Float16* qt   = (_Float16*)ws;
  _Float16* kt   = (_Float16*)(ws + qk_bytes);
  float*    vbuf = (float*)(ws + 2 * qk_bytes);
  float*    vout = out + (size_t)MASK_ELEMS;

  const int smem1 = (ICn * KP + KP * NP) * (int)sizeof(_Float16);                 // 156,672 B
  const int smem2 = (NP * ICn + ICn * NP) * (int)sizeof(_Float16)
                  + (NP * NP + 2 * NP) * (int)sizeof(float);                      // 227,968 B
  hipFuncSetAttribute((const void*)qkv_kernel,  hipFuncAttributeMaxDynamicSharedMemorySize, smem1);
  hipFuncSetAttribute((const void*)attn_kernel, hipFuncAttributeMaxDynamicSharedMemorySize, smem2);

  qkv_kernel<<<NT, 256, smem1, stream>>>(x, lam, index, Wq, bq, Wv, bv, qt, kt, vbuf, vout);
  attn_kernel<<<NT, 256, smem2, stream>>>(qt, kt, vbuf, out);
}
